// Attention_48163763258145
// MI455X (gfx1250) — matrix operand order repaired and checked by reading, not compiled, not run on hardware
//
#include <hip/hip_runtime.h>

#define LOG2E 1.4426950408889634f

typedef __attribute__((ext_vector_type(16))) __bf16        v16bf;
typedef __attribute__((ext_vector_type(8)))  float         v8f;
typedef __attribute__((ext_vector_type(4)))  unsigned int  u32x4;
typedef __attribute__((ext_vector_type(4)))  float         f32x4;

union Frag16 {
  v16bf v;
  u32x4 q[2];
  unsigned short h[16];
};

__device__ __forceinline__ unsigned short f2bf(float f) {
  unsigned int u = __float_as_uint(f);
  u += 0x7fffu + ((u >> 16) & 1u);
  return (unsigned short)(u >> 16);
}
__device__ __forceinline__ float bf2f(unsigned short h) {
  return __uint_as_float(((unsigned int)h) << 16);
}
// swap halves 0-15 <-> 16-31 (ds_swizzle, xor mask 0x10, and mask 0x1f)
__device__ __forceinline__ float xor16f(float v) {
  return __int_as_float(__builtin_amdgcn_ds_swizzle(__float_as_int(v), 0x401f));
}
__device__ __forceinline__ float bcastf(float v, int srcLane) {
  return __int_as_float(__builtin_amdgcn_ds_bpermute(srcLane << 2, __float_as_int(v)));
}

// ---------------------------------------------------------------- f32 -> bf16
__global__ __launch_bounds__(256) void cvt_bf16_kernel(
    const float* __restrict__ src, unsigned int* __restrict__ dst) {
  const int i = blockIdx.x * 256 + threadIdx.x;
  const f32x4* s4 = (const f32x4*)src;
  f32x4 a = s4[2 * i], b = s4[2 * i + 1];
  u32x4 o;
  o.x = f2bf(a.x) | ((unsigned)f2bf(a.y) << 16);
  o.y = f2bf(a.z) | ((unsigned)f2bf(a.w) << 16);
  o.z = f2bf(b.x) | ((unsigned)f2bf(b.y) << 16);
  o.w = f2bf(b.z) | ((unsigned)f2bf(b.w) << 16);
  ((u32x4*)dst)[i] = o;
}

__global__ __launch_bounds__(256) void cvt_bf16_small_kernel(
    const float* __restrict__ src, unsigned short* __restrict__ dst, int n) {
  const int i = blockIdx.x * 256 + threadIdx.x;
  if (i < n) dst[i] = f2bf(src[i]);
}

// ------------------------------------------------------- QKV projection GEMM
// C(16384,2304) = Xbf(16384,768) * Wbf(2304,768)^T + bias
// scatter: q/k -> (bh, s, 64) bf16 ; v -> transposed (bh, 64, s) bf16
__global__ __launch_bounds__(256) void qkv_gemm_kernel(
    const unsigned short* __restrict__ xbf,
    const unsigned short* __restrict__ wbf,
    const float* __restrict__ bias,
    unsigned short* __restrict__ qo,
    unsigned short* __restrict__ ko,
    unsigned short* __restrict__ vto) {
  const int lane = threadIdx.x & 31;
  const int wave = threadIdx.x >> 5;
  const int lr = lane & 15;
  const int hi = lane >> 4;
  const int row0 = blockIdx.y * 128 + (wave >> 1) * 32;  // 4 row-waves
  const int col0 = blockIdx.x * 128 + (wave & 1) * 64;   // 2 col-waves

  v8f acc[2][4] = {};
  for (int kk = 0; kk < 768; kk += 32) {
    Frag16 a[2], b[4];
#pragma unroll
    for (int i = 0; i < 2; i++) {
      const unsigned short* ap = xbf + (size_t)(row0 + i * 16 + lr) * 768 + kk + hi * 8;
      a[i].q[0] = *(const u32x4*)ap;
      a[i].q[1] = *(const u32x4*)(ap + 16);
    }
#pragma unroll
    for (int j = 0; j < 4; j++) {
      const unsigned short* bp = wbf + (size_t)(col0 + j * 16 + lr) * 768 + kk + hi * 8;
      b[j].q[0] = *(const u32x4*)bp;
      b[j].q[1] = *(const u32x4*)(bp + 16);
    }
#pragma unroll
    for (int i = 0; i < 2; i++)
#pragma unroll
      for (int j = 0; j < 4; j++)
        acc[i][j] = __builtin_amdgcn_wmma_f32_16x16x32_bf16(
            false, a[i].v, false, b[j].v, (short)0, acc[i][j], false, false);
  }
#pragma unroll
  for (int j = 0; j < 4; j++) {
    const int n = col0 + j * 16 + lr;
    const float bn = bias[n];
    const int which = n / 768;
    const int nn = n - which * 768;
    const int head = nn >> 6;
    const int c = nn & 63;
#pragma unroll
    for (int i = 0; i < 2; i++) {
#pragma unroll
      for (int r = 0; r < 8; r++) {
        const int m = row0 + i * 16 + hi * 8 + r;
        const int bidx = m >> 10;
        const int s = m & 1023;
        const int bh = bidx * 12 + head;
        const unsigned short val = f2bf(acc[i][j][r] + bn);
        if (which == 0)      qo[((size_t)bh << 16) + (s << 6) + c] = val;
        else if (which == 1) ko[((size_t)bh << 16) + (s << 6) + c] = val;
        else                 vto[((size_t)bh << 16) + ((size_t)c << 10) + s] = val;
      }
    }
  }
}

// --------------------------------------- decomposed rel-pos bias (WMMA GEMM)
// mode 0: rel_h[bh][s=g*32+w][k] = sum_c q[bh,s,c] * rposh[g-k+31][c]  (g = h)
// mode 1: rel_w[bh][s=h*32+g][k] = sum_c q[bh,s,c] * rposw[g-k+31][c]  (g = w)
// one wave per (bh, g): 32x32 output tile, K=64
__global__ __launch_bounds__(256) void relbias_wmma_kernel(
    const unsigned short* __restrict__ qbf,
    const unsigned short* __restrict__ rhbf,
    const unsigned short* __restrict__ rwbf,
    float* __restrict__ outh,
    float* __restrict__ outw) {
  const int lane = threadIdx.x & 31;
  const int wave = threadIdx.x >> 5;
  const int lr = lane & 15;
  const int hi = lane >> 4;
  const int mode = blockIdx.y;
  const int gt = blockIdx.x * 8 + wave;  // 192*32 waves
  const int bh = gt >> 5;
  const int g = gt & 31;
  const unsigned short* rp = mode ? rwbf : rhbf;
  float* out = mode ? outw : outh;

  v8f acc[2][2] = {};
#pragma unroll
  for (int ks = 0; ks < 2; ks++) {
    Frag16 a[2], b[2];
#pragma unroll
    for (int i = 0; i < 2; i++) {
      const int M = i * 16 + lr;
      const int s = mode ? (M * 32 + g) : (g * 32 + M);
      const unsigned short* ap =
          qbf + ((size_t)bh << 16) + ((size_t)s << 6) + ks * 32 + hi * 8;
      a[i].q[0] = *(const u32x4*)ap;
      a[i].q[1] = *(const u32x4*)(ap + 16);
    }
#pragma unroll
    for (int j = 0; j < 2; j++) {
      const int k = j * 16 + lr;
      const unsigned short* bp = rp + (size_t)(g - k + 31) * 64 + ks * 32 + hi * 8;
      b[j].q[0] = *(const u32x4*)bp;
      b[j].q[1] = *(const u32x4*)(bp + 16);
    }
#pragma unroll
    for (int i = 0; i < 2; i++)
#pragma unroll
      for (int j = 0; j < 2; j++)
        acc[i][j] = __builtin_amdgcn_wmma_f32_16x16x32_bf16(
            false, a[i].v, false, b[j].v, (short)0, acc[i][j], false, false);
  }
#pragma unroll
  for (int i = 0; i < 2; i++)
#pragma unroll
    for (int j = 0; j < 2; j++)
#pragma unroll
      for (int r = 0; r < 8; r++) {
        const int M = i * 16 + hi * 8 + r;
        const int s = mode ? (M * 32 + g) : (g * 32 + M);
        const int k = j * 16 + lr;
        out[((size_t)bh << 15) + s * 32 + k] = acc[i][j][r];
      }
}

// ------------------------------------------------------------ flash attention
// one wave per 16-row q tile; S^T = K*Q^T so score D-frag == P A-frag layout
__global__ __launch_bounds__(256) void attn_kernel(
    const unsigned short* __restrict__ qbf,
    const unsigned short* __restrict__ kbf,
    const unsigned short* __restrict__ vtbf,
    const float* __restrict__ relh,
    const float* __restrict__ relw,
    unsigned short* __restrict__ obf) {
  const int lane = threadIdx.x & 31;
  const int wave = threadIdx.x >> 5;
  const int lr = lane & 15;
  const int hi = lane >> 4;
  const int gt = blockIdx.x * 8 + wave;
  const int bh = gt >> 6;
  const int q0 = (gt & 63) << 4;
  const float SC = 0.125f * LOG2E;  // head_dim^-0.5 * log2(e)

  // Q as B-fragment of S^T = K * Q^T  (lane col = q row = lr)
  const unsigned short* qp = qbf + ((size_t)bh << 16) + ((size_t)(q0 + lr) << 6);
  Frag16 qb[2];
#pragma unroll
  for (int ks = 0; ks < 2; ks++) {
    const unsigned short* p = qp + ks * 32 + hi * 8;
    qb[ks].q[0] = *(const u32x4*)p;
    qb[ks].q[1] = *(const u32x4*)(p + 16);
  }
  // rel_w cached: key-in-chunk kw = t*16 + hi*8 + r (constant across chunks)
  float rw[2][8];
  {
    const float* rwp = relw + ((size_t)bh << 15) + ((size_t)(q0 + lr) << 5);
#pragma unroll
    for (int t = 0; t < 2; t++)
#pragma unroll
      for (int r = 0; r < 8; r++) rw[t][r] = rwp[t * 16 + hi * 8 + r] * LOG2E;
  }
  const float* rhp = relh + ((size_t)bh << 15) + ((size_t)(q0 + lr) << 5);

  float mrow = -1e30f, lsum = 0.f;
  v8f O[4] = {};
  const unsigned short* kbase = kbf + ((size_t)bh << 16);
  const unsigned short* vbase = vtbf + ((size_t)bh << 16);

  for (int ch = 0; ch < 32; ch++) {
    const int key0 = ch << 5;
    const float rh = rhp[ch] * LOG2E;  // k_h == chunk index: constant per chunk
    v8f st[2] = {};
#pragma unroll
    for (int t = 0; t < 2; t++) {
      const unsigned short* kp = kbase + ((size_t)(key0 + t * 16 + lr) << 6) + hi * 8;
#pragma unroll
      for (int ks = 0; ks < 2; ks++) {
        Frag16 a;
        a.q[0] = *(const u32x4*)(kp + ks * 32);
        a.q[1] = *(const u32x4*)(kp + ks * 32 + 16);
        st[t] = __builtin_amdgcn_wmma_f32_16x16x32_bf16(
            false, a.v, false, qb[ks].v, (short)0, st[t], false, false);
      }
    }
    // scores in exp2 domain; lane's q row = lr; its 16 keys = t*16 + hi*8 + r
    float sv[2][8];
    float mloc = -1e30f;
#pragma unroll
    for (int t = 0; t < 2; t++)
#pragma unroll
      for (int r = 0; r < 8; r++) {
        float s = st[t][r] * SC + rh + rw[t][r];
        sv[t][r] = s;
        mloc = fmaxf(mloc, s);
      }
    mloc = fmaxf(mloc, xor16f(mloc));      // other half holds the other 16 keys
    const float mnew = fmaxf(mrow, mloc);
    const float alpha = exp2f(mrow - mnew);
    float ps = 0.f;
    Frag16 pa;  // P A-fragment: element j -> key hi*8+j (t=0), 16+hi*8+j (t=1)
#pragma unroll
    for (int t = 0; t < 2; t++)
#pragma unroll
      for (int r = 0; r < 8; r++) {
        float p = exp2f(sv[t][r] - mnew);
        ps += p;
        pa.h[t * 8 + r] = f2bf(p);
      }
    lsum = lsum * alpha + ps;
    mrow = mnew;
    // rescale O accumulator rows (O row = r + 8*hi; alpha for row j is in lane j)
#pragma unroll
    for (int r = 0; r < 8; r++) {
      float ar = bcastf(alpha, r + 8 * hi);
#pragma unroll
      for (int ct = 0; ct < 4; ct++) O[ct][r] *= ar;
    }
    // O += P * V  (V transposed: lane col c = ct*16+lr, keys contiguous)
#pragma unroll
    for (int ct = 0; ct < 4; ct++) {
      Frag16 vb;
      const unsigned short* vp = vbase + ((size_t)(ct * 16 + lr) << 10) + key0 + hi * 8;
      vb.q[0] = *(const u32x4*)vp;
      vb.q[1] = *(const u32x4*)(vp + 16);
      O[ct] = __builtin_amdgcn_wmma_f32_16x16x32_bf16(
          false, pa.v, false, vb.v, (short)0, O[ct], false, false);
    }
  }
  const float ltot = lsum + xor16f(lsum);
  const float inv = 1.0f / ltot;
  const int b = bh / 12, head = bh - 12 * b;
#pragma unroll
  for (int r = 0; r < 8; r++) {
    float ir = bcastf(inv, r + 8 * hi);
    const int qq = q0 + r + 8 * hi;
    const size_t base = ((size_t)(b * 1024 + qq)) * 768 + head * 64;
#pragma unroll
    for (int ct = 0; ct < 4; ct++) obf[base + ct * 16 + lr] = f2bf(O[ct][r] * ir);
  }
}

// -------------------------------------------------------- output projection
__global__ __launch_bounds__(256) void proj_gemm_kernel(
    const unsigned short* __restrict__ abf,
    const unsigned short* __restrict__ wbf,
    const float* __restrict__ bias,
    float* __restrict__ out) {
  const int lane = threadIdx.x & 31;
  const int wave = threadIdx.x >> 5;
  const int lr = lane & 15;
  const int hi = lane >> 4;
  const int row0 = blockIdx.y * 128 + (wave >> 1) * 32;
  const int col0 = blockIdx.x * 128 + (wave & 1) * 64;

  v8f acc[2][4] = {};
  for (int kk = 0; kk < 768; kk += 32) {
    Frag16 a[2], b[4];
#pragma unroll
    for (int i = 0; i < 2; i++) {
      const unsigned short* ap = abf + (size_t)(row0 + i * 16 + lr) * 768 + kk + hi * 8;
      a[i].q[0] = *(const u32x4*)ap;
      a[i].q[1] = *(const u32x4*)(ap + 16);
    }
#pragma unroll
    for (int j = 0; j < 4; j++) {
      const unsigned short* bp = wbf + (size_t)(col0 + j * 16 + lr) * 768 + kk + hi * 8;
      b[j].q[0] = *(const u32x4*)bp;
      b[j].q[1] = *(const u32x4*)(bp + 16);
    }
#pragma unroll
    for (int i = 0; i < 2; i++)
#pragma unroll
      for (int j = 0; j < 4; j++)
        acc[i][j] = __builtin_amdgcn_wmma_f32_16x16x32_bf16(
            false, a[i].v, false, b[j].v, (short)0, acc[i][j], false, false);
  }
#pragma unroll
  for (int j = 0; j < 4; j++) {
    const int n = col0 + j * 16 + lr;
    const float bn = bias[n];
#pragma unroll
    for (int i = 0; i < 2; i++)
#pragma unroll
      for (int r = 0; r < 8; r++) {
        const int m = row0 + i * 16 + hi * 8 + r;
        out[(size_t)m * 768 + n] = acc[i][j][r] + bn;
      }
  }
}

extern "C" void kernel_launch(void* const* d_in, const int* in_sizes, int n_in,
                              void* d_out, int out_size, void* d_ws, size_t ws_size,
                              hipStream_t stream) {
  (void)in_sizes; (void)n_in; (void)out_size; (void)ws_size;
  const float* x      = (const float*)d_in[0];
  const float* w_qkv  = (const float*)d_in[1];
  const float* b_qkv  = (const float*)d_in[2];
  const float* w_proj = (const float*)d_in[3];
  const float* b_proj = (const float*)d_in[4];
  const float* rposh  = (const float*)d_in[5];
  const float* rposw  = (const float*)d_in[6];

  char* ws = (char*)d_ws;
  size_t off = 0;
  auto alloc = [&](size_t bytes) {
    void* p = ws + off;
    off += (bytes + 255) & ~(size_t)255;
    return p;
  };
  unsigned short* xbf   = (unsigned short*)alloc(16384ull * 768 * 2);
  unsigned short* wqbf  = (unsigned short*)alloc(2304ull * 768 * 2);
  unsigned short* wpbf  = (unsigned short*)alloc(768ull * 768 * 2);
  unsigned short* qb    = (unsigned short*)alloc(192ull * 1024 * 64 * 2);
  unsigned short* kb    = (unsigned short*)alloc(192ull * 1024 * 64 * 2);
  unsigned short* vtb   = (unsigned short*)alloc(192ull * 1024 * 64 * 2);
  float* relh           = (float*)alloc(192ull * 1024 * 32 * 4);
  float* relw           = (float*)alloc(192ull * 1024 * 32 * 4);
  unsigned short* rphbf = (unsigned short*)alloc(63ull * 64 * 2);
  unsigned short* rpwbf = (unsigned short*)alloc(63ull * 64 * 2);
  unsigned short* atb   = xbf;  // reuse: xbf dead after qkv_gemm (same size)

  cvt_bf16_kernel<<<16384 * 768 / 8 / 256, 256, 0, stream>>>(x, (unsigned int*)xbf);
  cvt_bf16_kernel<<<2304 * 768 / 8 / 256, 256, 0, stream>>>(w_qkv, (unsigned int*)wqbf);
  cvt_bf16_kernel<<<768 * 768 / 8 / 256, 256, 0, stream>>>(w_proj, (unsigned int*)wpbf);
  cvt_bf16_small_kernel<<<(63 * 64 + 255) / 256, 256, 0, stream>>>(rposh, rphbf, 63 * 64);
  cvt_bf16_small_kernel<<<(63 * 64 + 255) / 256, 256, 0, stream>>>(rposw, rpwbf, 63 * 64);

  qkv_gemm_kernel<<<dim3(2304 / 128, 16384 / 128), 256, 0, stream>>>(
      xbf, wqbf, b_qkv, qb, kb, vtb);

  relbias_wmma_kernel<<<dim3(192 * 32 / 8, 2), 256, 0, stream>>>(
      qb, rphbf, rpwbf, relh, relw);

  attn_kernel<<<192 * 64 / 8, 256, 0, stream>>>(qb, kb, vtb, relh, relw, atb);

  proj_gemm_kernel<<<dim3(768 / 128, 16384 / 128), 256, 0, stream>>>(
      atb, wpbf, b_proj, (float*)d_out);
}
